// KerasCrossAttentionModule_58720792871368
// MI455X (gfx1250) — matrix operand order repaired and checked by reading, not compiled, not run on hardware
//
#include <hip/hip_runtime.h>

typedef _Float16 h8   __attribute__((ext_vector_type(8)));
typedef _Float16 v16h __attribute__((ext_vector_type(16)));
typedef float    v8f  __attribute__((ext_vector_type(8)));
typedef int      vs4i __attribute__((vector_size(4 * sizeof(int))));

#define BATCH 8
#define SDIM  4096
#define CDIM  256
#define ATT_SCALE 0.0625f   // 256^-0.5
#define KT    64            // K-tile rows per iteration
#define NW    8             // waves per workgroup
#define QROWS (NW * 16)     // 128 Q rows per workgroup
#define KSTR  264           // padded LDS row stride (halves) for Q/K tiles
#define VSTR  72            // padded LDS row stride for V^T tile
#define PSTR  72            // padded LDS row stride for P scratch

#ifndef __has_builtin
#define __has_builtin(x) 0
#endif
#if __has_builtin(__builtin_amdgcn_global_load_async_to_lds_b128)
#define HAVE_ASYNC_LDS 1
#else
#define HAVE_ASYNC_LDS 0
#endif

// ---------------- preprocessing ----------------

// (B,C,S) NCHW -> (B,S,C) f16, adding pos[s,c]; scale folds softmax 1/sqrt(d) into Q
__global__ void prep_qk(const float* __restrict__ x, const float* __restrict__ pos,
                        _Float16* __restrict__ out, float scale) {
  __shared__ float tile[32][33];
  const int b = blockIdx.z, c0 = blockIdx.y * 32, s0 = blockIdx.x * 32;
  const int tx = threadIdx.x, ty = threadIdx.y;
  #pragma unroll
  for (int r = 0; r < 4; ++r) {
    int c = c0 + ty + r * 8;
    tile[ty + r * 8][tx] = x[((size_t)b * CDIM + c) * SDIM + s0 + tx];
  }
  __syncthreads();
  #pragma unroll
  for (int r = 0; r < 4; ++r) {
    int s = s0 + ty + r * 8;
    float v = (tile[tx][ty + r * 8] + pos[(size_t)s * CDIM + c0 + tx]) * scale;
    out[((size_t)b * SDIM + s) * CDIM + c0 + tx] = (_Float16)v;
  }
}

// values: NCHW (B,C,S) is already V^T[dv][k] layout -> elementwise f32->f16
__global__ void conv_v(const float* __restrict__ v, _Float16* __restrict__ o) {
  size_t i = ((size_t)blockIdx.x * blockDim.x + threadIdx.x) * 8;
  float4 a = *(const float4*)(v + i);
  float4 b = *(const float4*)(v + i + 4);
  h8 r;
  r[0] = (_Float16)a.x; r[1] = (_Float16)a.y; r[2] = (_Float16)a.z; r[3] = (_Float16)a.w;
  r[4] = (_Float16)b.x; r[5] = (_Float16)b.y; r[6] = (_Float16)b.z; r[7] = (_Float16)b.w;
  *(h8*)(o + i) = r;
}

// ---------------- async staging helpers ----------------

__device__ __forceinline__ void async_cp16(const _Float16* g, _Float16* l) {
#if HAVE_ASYNC_LDS
  __builtin_amdgcn_global_load_async_to_lds_b128((vs4i*)g, (vs4i*)l, 0, 0);
#else
  *(h8*)l = *(const h8*)g;
#endif
}

__device__ __forceinline__ void wait_async0() {
#if __has_builtin(__builtin_amdgcn_s_wait_asynccnt)
  __builtin_amdgcn_s_wait_asynccnt(0);
#else
  asm volatile("s_wait_asynccnt 0x0" ::: "memory");
#endif
}

// stage one 64x256 K tile and one 256x64 V^T tile into the given LDS buffers
__device__ __forceinline__ void stage_tiles(const _Float16* __restrict__ kg,
                                            const _Float16* __restrict__ vg,
                                            int kb, _Float16* sKb, _Float16* sVb,
                                            int tid) {
  {
    int row = tid >> 2, seg = tid & 3;   // coalesced: WG reads contiguous 32KB
    const _Float16* src = kg + (size_t)(kb + row) * CDIM + seg * 64;
    _Float16* dst = sKb + row * KSTR + seg * 64;
    #pragma unroll
    for (int i = 0; i < 8; ++i) async_cp16(src + i * 8, dst + i * 8);
  }
  {
    const _Float16* src = vg + (size_t)tid * SDIM + kb;  // 128B lines per dv row
    _Float16* dst = sVb + tid * VSTR;
    #pragma unroll
    for (int i = 0; i < 8; ++i) async_cp16(src + i * 8, dst + i * 8);
  }
}

// ---------------- flash attention ----------------
// Q tile = 128 rows (16 per wave), Q staged once in LDS, loop over 64-row K tiles
// with double-buffered async LDS staging. Online softmax is wave-local.

__launch_bounds__(256, 1)
__global__ void flash_attn(const _Float16* __restrict__ qh,
                           const _Float16* __restrict__ kh,
                           const _Float16* __restrict__ vt,
                           float* __restrict__ out) {
  __shared__ _Float16 sQ[QROWS * KSTR];       // Q tile, row-major [qrow][dqk]
  __shared__ _Float16 sK[2 * KT * KSTR];      // K tiles, row-major [krow][dqk]
  __shared__ _Float16 sV[2 * CDIM * VSTR];    // V^T tiles, [dv][krow]
  __shared__ _Float16 sP[NW * 16 * PSTR];     // per-wave P scratch [row][kcol]

  const int b     = blockIdx.y;
  const int qbase = blockIdx.x * QROWS;
  const int tid   = threadIdx.x;
  const int w = tid >> 5, lane = tid & 31, half = lane >> 4, l16 = lane & 15;

  const _Float16* kg = kh + (size_t)b * SDIM * CDIM;
  const _Float16* vg = vt + (size_t)b * CDIM * SDIM;

  // prologue: DMA Q tile (once) and K/V tile 0 into buffer 0
  {
    int row = tid >> 1, seg = tid & 1;   // 256B per thread, contiguous 64KB per WG
    const _Float16* src = qh + ((size_t)b * SDIM + qbase + row) * CDIM + seg * 128;
    _Float16* dst = sQ + row * KSTR + seg * 128;
    #pragma unroll
    for (int i = 0; i < 16; ++i) async_cp16(src + i * 8, dst + i * 8);
  }
  stage_tiles(kg, vg, 0, sK, sV, tid);

  v8f o_acc[16];
  v8f rm, rs;
  #pragma unroll
  for (int j = 0; j < 8; ++j) { rm[j] = -1e30f; rs[j] = 0.f; }
  #pragma unroll
  for (int i = 0; i < 16; ++i)
    #pragma unroll
    for (int j = 0; j < 8; ++j) o_acc[i][j] = 0.f;

  wait_async0();
  __syncthreads();

  const _Float16* qtile = sQ + (w * 16 + l16) * KSTR;  // this wave's A-frag rows

  for (int kt = 0; kt < SDIM / KT; ++kt) {
    const _Float16* sKb = sK + (kt & 1) * (KT * KSTR);
    const _Float16* sVb = sV + (kt & 1) * (CDIM * VSTR);

    // kick off DMA for next tile into the other buffer; overlaps all compute below
    if (kt + 1 < SDIM / KT)
      stage_tiles(kg, vg, (kt + 1) * KT, sK + ((kt + 1) & 1) * (KT * KSTR),
                  sV + ((kt + 1) & 1) * (CDIM * VSTR), tid);

    // ---- S = Q*K^T (scale pre-folded into Q). Wave owns rows 16w..16w+15, all 64 cols.
    v8f acc[4];
    #pragma unroll
    for (int nb = 0; nb < 4; ++nb)
      #pragma unroll
      for (int j = 0; j < 8; ++j) acc[nb][j] = 0.f;

    #pragma unroll
    for (int d = 0; d < 8; ++d) {
      // A-frag for this depth chunk: lanes<16 hold K {0..15}, lanes>=16 {16..31}
      v16h qa;
      {
        const _Float16* qp = qtile + d * 32 + half * 16;
        h8 lo = *(const h8*)qp, hi = *(const h8*)(qp + 8);
        #pragma unroll
        for (int i = 0; i < 8; ++i) { qa[i] = lo[i]; qa[i + 8] = hi[i]; }
      }
      v16h bf[4];                       // group loads so several stay in flight
      #pragma unroll
      for (int nb = 0; nb < 4; ++nb) {
        const _Float16* kp = sKb + (nb * 16 + l16) * KSTR + d * 32 + half * 16;
        h8 b0 = *(const h8*)kp, b1 = *(const h8*)(kp + 8);
        #pragma unroll
        for (int i = 0; i < 8; ++i) { bf[nb][i] = b0[i]; bf[nb][i + 8] = b1[i]; }
      }
      #pragma unroll
      for (int nb = 0; nb < 4; ++nb)
        acc[nb] = __builtin_amdgcn_wmma_f32_16x16x32_f16(
            false, qa, false, bf[nb], (short)0, acc[nb], false, false);
    }

    // ---- online softmax (C-layout: elem j = row j+8*half, lanes = columns)
    v8f t;
    #pragma unroll
    for (int j = 0; j < 8; ++j)
      t[j] = fmaxf(fmaxf(acc[0][j], acc[1][j]), fmaxf(acc[2][j], acc[3][j]));
    #pragma unroll
    for (int m = 1; m <= 8; m <<= 1)
      #pragma unroll
      for (int j = 0; j < 8; ++j) t[j] = fmaxf(t[j], __shfl_xor(t[j], m, 32));

    v8f sc;
    #pragma unroll
    for (int j = 0; j < 8; ++j) {
      float mn = fmaxf(rm[j], t[j]);
      sc[j] = __expf(rm[j] - mn);
      rm[j] = mn;
      rs[j] *= sc[j];
    }
    #pragma unroll
    for (int i = 0; i < 16; ++i)
      #pragma unroll
      for (int j = 0; j < 8; ++j) o_acc[i][j] *= sc[j];

    #pragma unroll
    for (int nb = 0; nb < 4; ++nb)
      #pragma unroll
      for (int j = 0; j < 8; ++j) acc[nb][j] = __expf(acc[nb][j] - rm[j]);

    v8f ps;
    #pragma unroll
    for (int j = 0; j < 8; ++j) ps[j] = (acc[0][j] + acc[1][j]) + (acc[2][j] + acc[3][j]);
    #pragma unroll
    for (int m = 1; m <= 8; m <<= 1)
      #pragma unroll
      for (int j = 0; j < 8; ++j) ps[j] += __shfl_xor(ps[j], m, 32);
    #pragma unroll
    for (int j = 0; j < 8; ++j) rs[j] += ps[j];

    // ---- P: C-layout -> LDS -> A-layout (per-wave private region; in-wave DS order)
    _Float16* pb = sP + w * 16 * PSTR;
    #pragma unroll
    for (int nb = 0; nb < 4; ++nb)
      #pragma unroll
      for (int j = 0; j < 8; ++j)
        pb[(j + half * 8) * PSTR + nb * 16 + l16] = (_Float16)acc[nb][j];
    asm volatile("s_wait_dscnt 0x0" ::: "memory");

    v16h pa[2];
    #pragma unroll
    for (int kc = 0; kc < 2; ++kc) {
      const _Float16* pp = pb + l16 * PSTR + kc * 32 + half * 16;
      h8 lo = *(const h8*)pp, hi = *(const h8*)(pp + 8);
      #pragma unroll
      for (int i = 0; i < 8; ++i) { pa[kc][i] = lo[i]; pa[kc][i + 8] = hi[i]; }
    }

    // ---- O += P * V  (B-frags from V^T: contiguous depth per dv lane); pairs of dv
    // blocks so 4 loads are in flight ahead of 4 WMMAs
    #pragma unroll
    for (int dv2 = 0; dv2 < 8; ++dv2) {
      v16h bf[4];
      #pragma unroll
      for (int u = 0; u < 2; ++u)
        #pragma unroll
        for (int kc = 0; kc < 2; ++kc) {
          const _Float16* vp =
              sVb + ((dv2 * 2 + u) * 16 + l16) * VSTR + kc * 32 + half * 16;
          h8 b0 = *(const h8*)vp, b1 = *(const h8*)(vp + 8);
          #pragma unroll
          for (int i = 0; i < 8; ++i) { bf[u*2+kc][i] = b0[i]; bf[u*2+kc][i + 8] = b1[i]; }
        }
      #pragma unroll
      for (int u = 0; u < 2; ++u)
        #pragma unroll
        for (int kc = 0; kc < 2; ++kc)
          o_acc[dv2 * 2 + u] = __builtin_amdgcn_wmma_f32_16x16x32_f16(
              false, pa[kc], false, bf[u * 2 + kc], (short)0, o_acc[dv2 * 2 + u],
              false, false);
    }

    // next tile's DMA must be complete and all waves done before buffer swap
    wait_async0();
    __syncthreads();
  }

  // ---- epilogue: out is NCHW (B, dv, s)
  v8f rinv;
  #pragma unroll
  for (int j = 0; j < 8; ++j) rinv[j] = __builtin_amdgcn_rcpf(rs[j]);
  float* op = out + (size_t)b * CDIM * SDIM;
  const int srow = qbase + w * 16 + half * 8;
  #pragma unroll
  for (int dvb = 0; dvb < 16; ++dvb) {
    const int dv = dvb * 16 + l16;
    #pragma unroll
    for (int j = 0; j < 8; ++j)
      op[(size_t)dv * SDIM + srow + j] = o_acc[dvb][j] * rinv[j];
  }
}

// ---------------- host launcher ----------------

extern "C" void kernel_launch(void* const* d_in, const int* in_sizes, int n_in,
                              void* d_out, int out_size, void* d_ws, size_t ws_size,
                              hipStream_t stream) {
  const float* q  = (const float*)d_in[0];
  const float* k  = (const float*)d_in[1];
  const float* v  = (const float*)d_in[2];
  const float* qp = (const float*)d_in[3];
  const float* kp = (const float*)d_in[4];

  const size_t elems = (size_t)BATCH * SDIM * CDIM;   // 8.4M
  _Float16* qh = (_Float16*)d_ws;                      // 16 MB
  _Float16* kh = qh + elems;                           // 16 MB
  _Float16* vt = kh + elems;                           // 16 MB

  dim3 tb(32, 8);
  dim3 tg(SDIM / 32, CDIM / 32, BATCH);
  prep_qk<<<tg, tb, 0, stream>>>(q, qp, qh, ATT_SCALE);
  prep_qk<<<tg, tb, 0, stream>>>(k, kp, kh, 1.0f);
  conv_v<<<dim3((unsigned)(elems / (256 * 8))), dim3(256), 0, stream>>>(v, vt);

  flash_attn<<<dim3(SDIM / QROWS, BATCH), dim3(256), 0, stream>>>(qh, kh, vt, (float*)d_out);
}
